// MLGraphConstruction_56349970923996
// MI455X (gfx1250) — compile-verified, matrix operand order checked by reading, not run on hardware
//
#include <hip/hip_runtime.h>
#include <hip/hip_fp16.h>

#define N_NODES 16384
#define D_IN    14
#define D_HID   128
#define D_EMB   8
#define KNN     64
#define ROWS    8            // query rows per workgroup in knn kernel
#define NTILES  (N_NODES / 16)

typedef _Float16 half16 __attribute__((ext_vector_type(16)));
typedef float    float8 __attribute__((ext_vector_type(8)));

// ---------------- Kernel 1: MLP embed + fp16 copy + squared norms ----------
__global__ void mlp_embed_kernel(const float* __restrict__ x,
                                 const float* __restrict__ W1,
                                 const float* __restrict__ b1,
                                 const float* __restrict__ W2,
                                 const float* __restrict__ b2,
                                 float* __restrict__ H,
                                 _Float16* __restrict__ H16,
                                 float* __restrict__ h2) {
    __shared__ float xr[D_IN];
    __shared__ float hid[D_HID];
    __shared__ float sq[D_EMB];
    const int row = blockIdx.x;
    const int t   = threadIdx.x;
    if (t < D_IN) xr[t] = x[row * D_IN + t];
    __syncthreads();
    float acc = b1[t];
    #pragma unroll
    for (int i = 0; i < D_IN; ++i) acc = fmaf(xr[i], W1[i * D_HID + t], acc);
    hid[t] = fmaxf(acc, 0.0f);
    __syncthreads();
    if (t < D_EMB) {
        float o = b2[t];
        for (int j = 0; j < D_HID; ++j) o = fmaf(hid[j], W2[j * D_EMB + t], o);
        H[row * D_EMB + t] = o;
        _Float16 o16 = (_Float16)o;
        H16[row * D_EMB + t] = o16;
        float of = (float)o16;
        sq[t] = of * of;
    }
    __syncthreads();
    if (t == 0) {
        float s = 0.0f;
        #pragma unroll
        for (int k = 0; k < D_EMB; ++k) s += sq[k];
        h2[row] = s;
    }
}

// ---------------- Kernel 2: fused WMMA distances + radix-select top-64 -----
__global__ void knn_kernel(const _Float16* __restrict__ H16,
                           const float* __restrict__ h2,
                           int* __restrict__ nbr) {
    extern __shared__ unsigned char smem[];
    // keys packed in row pairs: kp[r2][cand] = key(row 2*r2) | key(row 2*r2+1)<<16
    unsigned* kp   = (unsigned*)smem;                                   // 4*N u32
    unsigned* hist = (unsigned*)(smem + (ROWS / 2) * N_NODES * 4);      // ROWS*256 u32
    int* sel = (int*)(smem + (ROWS / 2) * N_NODES * 4 + ROWS * 256 * 4);
    // sel layout: [0..7]=hi bucket, [8..15]=count below hi bucket,
    //             [16..23]=threshold key T, [24..31]=strict-less count L

    const int tid   = threadIdx.x;
    const int lane  = tid & 31;
    const int wave  = tid >> 5;
    const int qbase = blockIdx.x * ROWS;

    // ---- Phase 1: WMMA Gram tiles -> quantized distance keys in LDS ----
    union Frag { half16 h; unsigned u[8]; } fa;
    #pragma unroll
    for (int i = 0; i < 8; ++i) fa.u[i] = 0u;
    if (lane < ROWS) {
        const uint4 v = *(const uint4*)(H16 + (qbase + lane) * D_EMB);
        fa.u[0] = v.x; fa.u[1] = v.y; fa.u[2] = v.z; fa.u[3] = v.w;
    }
    float h2q[ROWS];
    #pragma unroll
    for (int r = 0; r < ROWS; ++r) h2q[r] = h2[qbase + r];   // WG-uniform -> SMEM loads

    const int cl = lane & 15;         // clamped candidate-row lane
    const bool lo = (lane < 16);

    // software-pipelined candidate loads (double buffer)
    uint4 vraw; float hraw;
    {
        const int rowi = wave * 16 + cl;
        vraw = *(const uint4*)(H16 + rowi * D_EMB);
        hraw = h2[rowi];
    }
    for (int tile = wave; tile < NTILES; tile += 8) {
        uint4 vnext = vraw; float hnext = hraw;
        if (tile + 8 < NTILES) {                     // wave-uniform branch
            const int rowi = (tile + 8) * 16 + cl;
            vnext = *(const uint4*)(H16 + rowi * D_EMB);
            hnext = h2[rowi];
        }
        Frag fb;
        fb.u[0] = lo ? vraw.x : 0u; fb.u[1] = lo ? vraw.y : 0u;
        fb.u[2] = lo ? vraw.z : 0u; fb.u[3] = lo ? vraw.w : 0u;
        fb.u[4] = 0u; fb.u[5] = 0u; fb.u[6] = 0u; fb.u[7] = 0u;
        const float h2c = lo ? hraw : 0.0f;

        float8 c = {0.f, 0.f, 0.f, 0.f, 0.f, 0.f, 0.f, 0.f};
        float8 d = __builtin_amdgcn_wmma_f32_16x16x32_f16(
            false, fa.h, false, fb.h, (short)0, c, false, false);

        if (lo) {
            const int cand = tile * 16 + lane;
            #pragma unroll
            for (int r2 = 0; r2 < ROWS / 2; ++r2) {
                float d2a = fmaxf(h2q[2 * r2]     + h2c - 2.0f * d[2 * r2],     0.0f);
                float d2b = fmaxf(h2q[2 * r2 + 1] + h2c - 2.0f * d[2 * r2 + 1], 0.0f);
                unsigned k0 = __float_as_uint(d2a) >> 16;    // monotone 16-bit key
                unsigned k1 = __float_as_uint(d2b) >> 16;
                if (cand == qbase + 2 * r2)     k0 = 0xFFFFu;  // self -> +inf
                if (cand == qbase + 2 * r2 + 1) k1 = 0xFFFFu;
                kp[r2 * N_NODES + cand] = k0 | (k1 << 16);
            }
        }
        vraw = vnext; hraw = hnext;
    }
    __syncthreads();

    const unsigned* kr = kp + (wave >> 1) * N_NODES;
    const int ksh = (wave & 1) * 16;                 // wave-uniform half select

    // ---- Phase 2: high-byte histogram + bucket of the 64th smallest ----
    for (int i = tid; i < ROWS * 256; i += 256) hist[i] = 0u;
    __syncthreads();
    {
        unsigned* h = hist + wave * 256;
        for (int i = lane; i < N_NODES; i += 32) {
            unsigned k = (kr[i] >> ksh) & 0xFFFFu;
            atomicAdd(&h[k >> 8], 1u);
        }
    }
    __syncthreads();
    if (tid < ROWS) {
        unsigned cum = 0; int bsel = 255; unsigned lh = 0;
        for (int b = 0; b < 256; ++b) {
            unsigned c = hist[tid * 256 + b];
            if (cum + c >= KNN) { bsel = b; lh = cum; break; }
            cum += c;
        }
        sel[tid] = bsel; sel[8 + tid] = (int)lh;
    }
    __syncthreads();

    // ---- Phase 3: low-byte refinement -> exact threshold key T, count L ----
    for (int i = tid; i < ROWS * 256; i += 256) hist[i] = 0u;
    __syncthreads();
    {
        const int b1v = sel[wave];
        unsigned* h = hist + wave * 256;
        for (int i = lane; i < N_NODES; i += 32) {
            unsigned k = (kr[i] >> ksh) & 0xFFFFu;
            if ((int)(k >> 8) == b1v) atomicAdd(&h[k & 255u], 1u);
        }
    }
    __syncthreads();
    if (tid < ROWS) {
        unsigned cum = (unsigned)sel[8 + tid];
        const int b1v = sel[tid];
        int Tv = (b1v << 8) | 255; unsigned Lv = cum;
        for (int b = 0; b < 256; ++b) {
            unsigned c = hist[tid * 256 + b];
            if (cum + c >= KNN) { Tv = (b1v << 8) | b; Lv = cum; break; }
            cum += c;
        }
        sel[16 + tid] = Tv; sel[24 + tid] = (int)Lv;
    }
    __syncthreads();

    // ---- Phase 4: deterministic index-ordered emission (wave per row) ----
    {
        const unsigned Tv = (unsigned)sel[16 + wave];
        const int Lv = sel[24 + wave];
        int* out = nbr + (qbase + wave) * KNN;
        int nLess = 0, nTie = 0;
        const unsigned lower = (1u << lane) - 1u;
        for (int i = lane; i < N_NODES; i += 32) {
            const unsigned k = (kr[i] >> ksh) & 0xFFFFu;
            const bool less = (k < Tv);
            const bool tie  = (k == Tv);
            const unsigned mL = __builtin_amdgcn_ballot_w32(less);
            const unsigned mT = __builtin_amdgcn_ballot_w32(tie);
            if (less) out[nLess + __popc(mL & lower)] = i;
            if (tie) {
                int s = Lv + nTie + __popc(mT & lower);
                if (s < KNN) out[s] = i;
            }
            nLess += __popc(mL);
            nTie  += __popc(mT);
        }
    }
}

// ---------------- Kernel 3: edge masks, labels, features (coalesced) -------
__global__ void edge_kernel(const float* __restrict__ x,
                            const int* __restrict__ pid,
                            const float* __restrict__ H,
                            const int* __restrict__ nbr,
                            int* __restrict__ ei_src,
                            int* __restrict__ ei_dst,
                            int* __restrict__ y,
                            int* __restrict__ mask_out,
                            float* __restrict__ ef) {
    __shared__ float fs[128 * 2 * D_IN];
    const int t = threadIdx.x;
    const int e = blockIdx.x * 128 + t;
    const int dst = e >> 6;
    const int src = nbr[e] & (N_NODES - 1);   // bound defensively

    float d2 = 0.0f;
    #pragma unroll
    for (int k = 0; k < D_EMB; ++k) {
        float diff = H[src * D_EMB + k] - H[dst * D_EMB + k];
        d2 = fmaf(diff, diff, d2);
    }
    const int m = (sqrtf(d2) < 1.0f) ? 1 : 0;
    ei_src[e] = src;
    ei_dst[e] = dst;
    y[e] = (m && (pid[src] == pid[dst])) ? 1 : 0;
    mask_out[e] = m;

    const float fm = (float)m;
    #pragma unroll
    for (int k = 0; k < D_IN; ++k) {
        const float a = x[src * D_IN + k];
        const float b = x[dst * D_IN + k];
        fs[t * (2 * D_IN) + k]        = (a - b) * fm;
        fs[t * (2 * D_IN) + D_IN + k] = (a + b) * fm;
    }
    __syncthreads();
    float* dp = ef + (size_t)(blockIdx.x) * 128 * 2 * D_IN;
    for (int i = t; i < 128 * 2 * D_IN; i += 128) dp[i] = fs[i];
}

extern "C" void kernel_launch(void* const* d_in, const int* in_sizes, int n_in,
                              void* d_out, int out_size, void* d_ws, size_t ws_size,
                              hipStream_t stream) {
    (void)in_sizes; (void)n_in; (void)out_size; (void)ws_size;
    const float* x  = (const float*)d_in[0];
    const int* pid  = (const int*)d_in[1];
    const float* W1 = (const float*)d_in[2];
    const float* b1 = (const float*)d_in[3];
    const float* W2 = (const float*)d_in[4];
    const float* b2 = (const float*)d_in[5];

    // outputs, flat in return order: H, edge_index[2,E], y, edge_mask, edge_features
    const int E = N_NODES * KNN;
    float* out   = (float*)d_out;
    float* H     = out;                            // N*8
    int* ei_src  = (int*)(out + N_NODES * D_EMB);  // E
    int* ei_dst  = ei_src + E;                     // E
    int* y       = ei_dst + E;                     // E
    int* msk     = y + E;                          // E
    float* ef    = (float*)(msk + E);              // E*28

    unsigned char* ws = (unsigned char*)d_ws;
    _Float16* H16 = (_Float16*)ws;                                   // N*8*2 B
    float* h2     = (float*)(ws + N_NODES * D_EMB * 2);              // N*4 B
    int* nbr      = (int*)(ws + N_NODES * D_EMB * 2 + N_NODES * 4);  // N*64*4 B

    mlp_embed_kernel<<<N_NODES, 128, 0, stream>>>(x, W1, b1, W2, b2, H, H16, h2);

    const size_t lds = (size_t)(ROWS / 2) * N_NODES * 4 + (size_t)ROWS * 256 * 4 + 32 * 4;
    hipFuncSetAttribute(reinterpret_cast<const void*>(knn_kernel),
                        hipFuncAttributeMaxDynamicSharedMemorySize, (int)lds);
    knn_kernel<<<N_NODES / ROWS, 256, lds, stream>>>(H16, h2, nbr);

    edge_kernel<<<E / 128, 128, 0, stream>>>(x, pid, H, nbr,
                                             ei_src, ei_dst, y, msk, ef);
}
